// VisionMamba_18159121727859
// MI455X (gfx1250) — compile-verified, matrix operand order checked by reading, not run on hardware
//
#include <hip/hip_runtime.h>
#include <stdint.h>

// ---------------- model constants ----------------
#define DEPTH 24
#define DMODEL 192
#define DI 384
#define NSTATE 16
#define DTR 12
#define KCONV 4
#define BATCH 64
#define LTOK 196
#define NTOK (BATCH * LTOK)        // 12544
#define XPN (DTR + 2 * NSTATE)     // 44
#define NCLS 1000
#define PKK 768                    // 3*16*16 patch im2col K

// ---------------- CDNA5 types ----------------
typedef __attribute__((ext_vector_type(16))) __bf16 v16bf;
typedef __attribute__((ext_vector_type(8)))  float  v8f;
typedef __attribute__((ext_vector_type(4)))  unsigned int u32x4;
typedef __attribute__((ext_vector_type(8)))  int    i32x8;
typedef __attribute__((ext_vector_type(4)))  int    i32x4;

static __device__ __forceinline__ __bf16 f2bf(float f) {
  unsigned u = __builtin_bit_cast(unsigned, f);
  u += 0x7FFFu + ((u >> 16) & 1u);               // round-to-nearest-even
  unsigned short h = (unsigned short)(u >> 16);
  return __builtin_bit_cast(__bf16, h);
}

static __device__ __forceinline__ float siluf(float x) {
  return x / (1.0f + __expf(-x));
}

// ---------------- f32 -> bf16 bulk convert ----------------
__global__ void f2bf_kernel(const float* __restrict__ in, __bf16* __restrict__ out, int n) {
  int i = blockIdx.x * blockDim.x + threadIdx.x;
  if (i < n) out[i] = f2bf(in[i]);
}

// ---------------- zero fill ----------------
__global__ void zero_kernel(float* __restrict__ p, int n) {
  int i = blockIdx.x * blockDim.x + threadIdx.x;
  if (i < n) p[i] = 0.0f;
}

// ---------------- im2col for 16x16 stride-16 patches ----------------
__global__ void im2col_kernel(const float* __restrict__ x, __bf16* __restrict__ A0) {
  int idx = blockIdx.x * blockDim.x + threadIdx.x;
  if (idx >= NTOK * PKK) return;
  int m = idx / PKK, k = idx - m * PKK;
  int b = m / LTOK, l = m - b * LTOK;
  int c = k >> 8, r = k & 255;
  int py = r >> 4, px = r & 15;
  int iy = (l / 14) * 16 + py;
  int ix = (l % 14) * 16 + px;
  float v = x[(((size_t)b * 3 + c) * 224 + iy) * 224 + ix];
  A0[idx] = f2bf(v);
}

// ---------------- WMMA GEMM:  C[M,N] = A[M,K] * W[N,K]^T (+bias) ----------------
// A, W bf16 row-major; C f32 row-major. K must be a multiple of 32.
// Block: 256 threads = 8 waves; wave grid 4x2; each wave owns a 32x32 C tile
// (2x2 v_wmma_f32_16x16x32_bf16 with full A/B fragment reuse) of a 128x64 block tile.
#define BM 128
#define BN 64
#define BK 32
__global__ __launch_bounds__(256)
void gemm_bf16_wmma(const __bf16* __restrict__ A, const __bf16* __restrict__ W,
                    const float* __restrict__ bias, float* __restrict__ C,
                    int M, int N, int K) {
  __shared__ __bf16 As[2][BM][BK];
  __shared__ __bf16 Bs[2][BN][BK];

  const int tid  = threadIdx.x;
  const int wave = tid >> 5;        // 0..7
  const int lane = tid & 31;
  const int wm   = wave >> 1;       // 0..3  (32-row slice)
  const int wn   = wave & 1;        // 0..1  (32-col slice)
  const int bm   = blockIdx.y * BM;
  const int bn   = blockIdx.x * BN;

  // cooperative tile loads: 16B (8 x bf16) per access, 4 threads per 32-elem row
  const int arow = tid >> 2;        // 0..63
  const int acol = (tid & 3) * 8;   // 0,8,16,24

  auto load_tiles = [&](int kk, int buf) {
#pragma unroll
    for (int p = 0; p < 2; ++p) {           // A: 128 rows in two passes
      int r = p * 64 + arow;
      int gr = bm + r;
      u32x4 v = {0, 0, 0, 0};
      if (gr < M) v = *(const u32x4*)&A[(size_t)gr * K + kk + acol];
      *(u32x4*)&As[buf][r][acol] = v;
    }
    {                                        // B: 64 rows in one pass
      int gr = bn + arow;
      u32x4 v = {0, 0, 0, 0};
      if (gr < N) v = *(const u32x4*)&W[(size_t)gr * K + kk + acol];
      *(u32x4*)&Bs[buf][arow][acol] = v;
    }
  };

  v8f acc[2][2] = {};
  load_tiles(0, 0);
  __syncthreads();

  const int frow = lane & 15;
  const int kh   = (lane >> 4) * 16;

  for (int kk = 0; kk < BK * ((K + BK - 1) / BK); kk += BK) {
    int buf = (kk / BK) & 1;
    if (kk + BK < K) {
      if (bm + arow < M) __builtin_prefetch(&A[(size_t)(bm + arow) * K + kk + BK + acol], 0, 1);
      load_tiles(kk + BK, buf ^ 1);
    }
    v16bf a0 = *(const v16bf*)&As[buf][wm * 32 + frow][kh];
    v16bf a1 = *(const v16bf*)&As[buf][wm * 32 + 16 + frow][kh];
    v16bf b0 = *(const v16bf*)&Bs[buf][wn * 32 + frow][kh];
    v16bf b1 = *(const v16bf*)&Bs[buf][wn * 32 + 16 + frow][kh];
    acc[0][0] = __builtin_amdgcn_wmma_f32_16x16x32_bf16(false, a0, false, b0, (short)0, acc[0][0], false, false);
    acc[0][1] = __builtin_amdgcn_wmma_f32_16x16x32_bf16(false, a0, false, b1, (short)0, acc[0][1], false, false);
    acc[1][0] = __builtin_amdgcn_wmma_f32_16x16x32_bf16(false, a1, false, b0, (short)0, acc[1][0], false, false);
    acc[1][1] = __builtin_amdgcn_wmma_f32_16x16x32_bf16(false, a1, false, b1, (short)0, acc[1][1], false, false);
    __syncthreads();
  }

  // C layout: VGPR j -> M = j (lanes 0-15) / 8+j (lanes 16-31); N = lane&15
  const int n  = lane & 15;
  const int mb = (lane >> 4) * 8;
#pragma unroll
  for (int tj = 0; tj < 2; ++tj) {
    int gn = bn + wn * 32 + tj * 16 + n;
    float bv = 0.0f;
    if (bias != nullptr && gn < N) bv = bias[gn];
#pragma unroll
    for (int ti = 0; ti < 2; ++ti) {
#pragma unroll
      for (int j = 0; j < 8; ++j) {
        int gm = bm + wm * 32 + ti * 16 + mb + j;
        if (gm < M && gn < N) C[(size_t)gm * N + gn] = acc[ti][tj][j] + bv;
      }
    }
  }
}

// ---------------- fused residual-add + LayerNorm (wave per token) ----------------
__global__ __launch_bounds__(256)
void add_ln_kernel(float* __restrict__ res, const float* __restrict__ hid,
                   const float* __restrict__ w, const float* __restrict__ b,
                   __bf16* __restrict__ out_bf, int ntok) {
  int wave = threadIdx.x >> 5, lane = threadIdx.x & 31;
  int tok = blockIdx.x * 8 + wave;
  if (tok >= ntok) return;
  size_t base = (size_t)tok * DMODEL;
  float v[6];
  float sum = 0.0f;
#pragma unroll
  for (int i = 0; i < 6; ++i) {
    int c = i * 32 + lane;
    float t = res[base + c] + hid[base + c];
    v[i] = t; sum += t;
  }
#pragma unroll
  for (int off = 16; off > 0; off >>= 1) sum += __shfl_xor(sum, off, 32);
  float mu = sum * (1.0f / DMODEL);
  float sq = 0.0f;
#pragma unroll
  for (int i = 0; i < 6; ++i) { float d = v[i] - mu; sq += d * d; }
#pragma unroll
  for (int off = 16; off > 0; off >>= 1) sq += __shfl_xor(sq, off, 32);
  float rstd = rsqrtf(sq * (1.0f / DMODEL) + 1e-5f);
#pragma unroll
  for (int i = 0; i < 6; ++i) {
    int c = i * 32 + lane;
    res[base + c] = v[i];                          // residual update
    out_bf[base + c] = f2bf((v[i] - mu) * rstd * w[c] + b[c]);
  }
}

// ---------------- depthwise causal conv1d(K=4) + SiLU ----------------
__global__ __launch_bounds__(256)
void conv_silu_kernel(const float* __restrict__ xz, const float* __restrict__ cw,
                      const float* __restrict__ cb, float* __restrict__ xc,
                      __bf16* __restrict__ xcbf) {
  int idx = blockIdx.x * blockDim.x + threadIdx.x;
  if (idx >= NTOK * DI) return;
  int tok = idx / DI, e = idx - tok * DI;
  int b = tok / LTOK, l = tok - b * LTOK;
  float s = cb[e];
#pragma unroll
  for (int k = 0; k < KCONV; ++k) {
    int ls = l + k - (KCONV - 1);
    if (ls >= 0)
      s += xz[((size_t)(b * LTOK + ls)) * (2 * DI) + e] * cw[e * KCONV + k];
  }
  float v = siluf(s);
  xc[idx] = v;
  xcbf[idx] = f2bf(v);
}

// ---------------- fused selective scan ----------------
// Grid (3, 64): block owns one batch element, 128 channels. The shared 44-float
// xdbl row (dt/B/C) is staged once per timestep in LDS instead of per lane.
__global__ __launch_bounds__(128)
void scan_kernel(const float* __restrict__ xdbl,  // (NTOK, 44)
                 const float* __restrict__ xc,    // (NTOK, DI)
                 const float* __restrict__ xz,    // (NTOK, 2*DI), z at col DI
                 const float* __restrict__ dt_w,  // (DI, DTR)
                 const float* __restrict__ dt_b,  // (DI)
                 const float* __restrict__ A_log, // (DI, N)
                 const float* __restrict__ Dp,    // (DI)
                 __bf16* __restrict__ ybf) {      // (NTOK, DI)
  __shared__ float srow[XPN];
  const int e = blockIdx.x * 128 + threadIdx.x;
  const int b = blockIdx.y;

  float wdt[DTR];
#pragma unroll
  for (int r = 0; r < DTR; ++r) wdt[r] = dt_w[e * DTR + r];
  float db = dt_b[e], dpe = Dp[e];
  float a[NSTATE], h[NSTATE];
#pragma unroll
  for (int n = 0; n < NSTATE; ++n) { a[n] = -__expf(A_log[e * NSTATE + n]); h[n] = 0.0f; }

  for (int l = 0; l < LTOK; ++l) {
    int tok = b * LTOK + l;
    if (threadIdx.x < XPN) srow[threadIdx.x] = xdbl[(size_t)tok * XPN + threadIdx.x];
    __syncthreads();

    float dtv = db;
#pragma unroll
    for (int r = 0; r < DTR; ++r) dtv += srow[r] * wdt[r];
    float delta = log1pf(__expf(dtv));             // softplus
    float xcv = xc[(size_t)tok * DI + e];
    float zv  = xz[(size_t)tok * (2 * DI) + DI + e];
    float dbx = delta * xcv;
    float y = dpe * xcv;
#pragma unroll
    for (int n = 0; n < NSTATE; ++n) {
      float dA = __expf(delta * a[n]);
      h[n] = dA * h[n] + dbx * srow[DTR + n];
      y += h[n] * srow[DTR + NSTATE + n];
    }
    y *= siluf(zv);
    ybf[(size_t)tok * DI + e] = f2bf(y);
    __syncthreads();
  }
}

// ---------------- final: (res+hid) on last token, LayerNorm, emit bf16 ----------------
__global__ __launch_bounds__(256)
void final_ln_kernel(const float* __restrict__ res, const float* __restrict__ hid,
                     const float* __restrict__ w, const float* __restrict__ b,
                     __bf16* __restrict__ out_bf) {
  int wave = threadIdx.x >> 5, lane = threadIdx.x & 31;
  int t = blockIdx.x * 8 + wave;
  if (t >= BATCH) return;
  size_t base = ((size_t)t * LTOK + (LTOK - 1)) * DMODEL;
  float v[6];
  float sum = 0.0f;
#pragma unroll
  for (int i = 0; i < 6; ++i) {
    int c = i * 32 + lane;
    float x = res[base + c] + hid[base + c];
    v[i] = x; sum += x;
  }
#pragma unroll
  for (int off = 16; off > 0; off >>= 1) sum += __shfl_xor(sum, off, 32);
  float mu = sum * (1.0f / DMODEL);
  float sq = 0.0f;
#pragma unroll
  for (int i = 0; i < 6; ++i) { float d = v[i] - mu; sq += d * d; }
#pragma unroll
  for (int off = 16; off > 0; off >>= 1) sq += __shfl_xor(sq, off, 32);
  float rstd = rsqrtf(sq * (1.0f / DMODEL) + 1e-5f);
#pragma unroll
  for (int i = 0; i < 6; ++i) {
    int c = i * 32 + lane;
    out_bf[(size_t)t * DMODEL + c] = f2bf((v[i] - mu) * rstd * w[c] + b[c]);
  }
}

// ---------------- Tensor Data Mover warm-up (async tensor path) ----------------
// Pulls a 64x32 bf16 tile of the layer's in_proj weights into LDS via TDM,
// waits on TENSORcnt, and sinks one element (deterministic L2 warm).
#if __clang_major__ >= 23
#define TDM_LOAD(g0, g1, g2, g3) \
  __builtin_amdgcn_tensor_load_to_lds(g0, g1, g2, g3, (i32x8){0,0,0,0,0,0,0,0}, 0)
#else
#define TDM_LOAD(g0, g1, g2, g3) \
  __builtin_amdgcn_tensor_load_to_lds(g0, g1, g2, g3, 0)
#endif

__global__ void tdm_warm_kernel(const __bf16* __restrict__ src, float* __restrict__ sink) {
  __shared__ __bf16 buf[BN * BK];
  unsigned long long ga = (unsigned long long)(uintptr_t)src;
  u32x4 g0;
  g0[0] = 1u;                                        // count=1 (valid user D#)
  g0[1] = 0u;                                        // lds_addr = 0
  g0[2] = (unsigned)(ga & 0xFFFFFFFFu);              // global_addr[31:0]
  g0[3] = (unsigned)((ga >> 32) & 0x1FFFFFFu) | (2u << 30);  // addr[56:32] | type=2
  i32x8 g1;
  g1[0] = (1 << 16);                                 // data_size=2B; wg_mask=0
  g1[1] = (DMODEL & 0xFFFF) << 16;                   // tensor_dim0 (K=192) @ bits 79:48
  g1[2] = ((2 * DI) & 0xFFFF) << 16;                 // tensor_dim1 (768) @ bits 111:80
  g1[3] = (BK << 16);                                // tile_dim0=32 @ bits 127:112
  g1[4] = BN;                                        // tile_dim1=64
  g1[5] = DMODEL;                                    // tensor_dim0_stride=192
  g1[6] = 0;
  g1[7] = 0;
  i32x4 z4 = {0, 0, 0, 0};
  TDM_LOAD(g0, g1, z4, z4);
  __builtin_amdgcn_s_wait_tensorcnt(0);
  __builtin_amdgcn_s_cluster_barrier();              // NOP outside a cluster
  __syncthreads();
  if (threadIdx.x == 0) sink[0] = (float)buf[0];
}

// ================= host side =================
static inline size_t alignup(size_t x) { return (x + 255) & ~(size_t)255; }

extern "C" void kernel_launch(void* const* d_in, const int* in_sizes, int n_in,
                              void* d_out, int out_size, void* d_ws, size_t ws_size,
                              hipStream_t stream) {
  (void)in_sizes; (void)n_in; (void)out_size; (void)ws_size;
  const float* x       = (const float*)d_in[0];
  const float* patch_w = (const float*)d_in[1];
  const float* patch_b = (const float*)d_in[2];
  const float* ln_w    = (const float*)d_in[3];
  const float* ln_b    = (const float*)d_in[4];
  const float* in_w    = (const float*)d_in[5];
  const float* conv_w  = (const float*)d_in[6];
  const float* conv_b  = (const float*)d_in[7];
  const float* xproj_w = (const float*)d_in[8];
  const float* dt_w    = (const float*)d_in[9];
  const float* dt_b    = (const float*)d_in[10];
  const float* A_log   = (const float*)d_in[11];
  const float* D_param = (const float*)d_in[12];
  const float* out_w   = (const float*)d_in[13];
  const float* normf_w = (const float*)d_in[14];
  const float* normf_b = (const float*)d_in[15];
  const float* head_w  = (const float*)d_in[16];
  const float* head_b  = (const float*)d_in[17];
  float* out = (float*)d_out;

  // -------- workspace carve-up --------
  char* p = (char*)d_ws;
  auto take = [&](size_t bytes) { char* r = p; p += alignup(bytes); return r; };
  float*  ws_res   = (float*)take((size_t)NTOK * DMODEL * 4);
  float*  ws_hid   = (float*)take((size_t)NTOK * DMODEL * 4);
  __bf16* ws_hnbf  = (__bf16*)take((size_t)NTOK * DMODEL * 2);
  float*  ws_xz    = (float*)take((size_t)NTOK * 2 * DI * 4);
  float*  ws_xc    = (float*)take((size_t)NTOK * DI * 4);
  __bf16* ws_xcbf  = (__bf16*)take((size_t)NTOK * DI * 2);
  float*  ws_xdbl  = (float*)take((size_t)NTOK * XPN * 4);
  __bf16* ws_ybf   = (__bf16*)take((size_t)NTOK * DI * 2);
  __bf16* ws_A0    = (__bf16*)take((size_t)NTOK * PKK * 2);
  __bf16* wb_in    = (__bf16*)take((size_t)DEPTH * 2 * DI * DMODEL * 2);
  __bf16* wb_xp    = (__bf16*)take((size_t)DEPTH * XPN * DI * 2);
  __bf16* wb_out   = (__bf16*)take((size_t)DEPTH * DMODEL * DI * 2);
  __bf16* wb_patch = (__bf16*)take((size_t)DMODEL * PKK * 2);
  __bf16* wb_head  = (__bf16*)take((size_t)NCLS * DMODEL * 2);
  __bf16* ws_hd_in = (__bf16*)take((size_t)BATCH * DMODEL * 2);
  float*  tdm_sink = (float*)take((size_t)DEPTH * 4);

  auto cvt = [&](const float* src, __bf16* dst, int n) {
    f2bf_kernel<<<(n + 255) / 256, 256, 0, stream>>>(src, dst, n);
  };
  auto gemm = [&](const __bf16* A, const __bf16* W, const float* bias,
                  float* C, int M, int N, int K) {
    dim3 g((N + BN - 1) / BN, (M + BM - 1) / BM);
    gemm_bf16_wmma<<<g, 256, 0, stream>>>(A, W, bias, C, M, N, K);
  };

  // -------- weights -> bf16 (stateless: every call) --------
  cvt(in_w,    wb_in,    DEPTH * 2 * DI * DMODEL);
  cvt(xproj_w, wb_xp,    DEPTH * XPN * DI);
  cvt(out_w,   wb_out,   DEPTH * DMODEL * DI);
  cvt(patch_w, wb_patch, DMODEL * PKK);
  cvt(head_w,  wb_head,  NCLS * DMODEL);

  // -------- patch embedding --------
  im2col_kernel<<<(NTOK * PKK + 255) / 256, 256, 0, stream>>>(x, ws_A0);
  gemm(ws_A0, wb_patch, patch_b, ws_hid, NTOK, DMODEL, PKK);
  zero_kernel<<<(NTOK * DMODEL + 255) / 256, 256, 0, stream>>>(ws_res, NTOK * DMODEL);

  // -------- 24 Mamba blocks --------
  for (int lyr = 0; lyr < DEPTH; ++lyr) {
    add_ln_kernel<<<(NTOK + 7) / 8, 256, 0, stream>>>(
        ws_res, ws_hid, ln_w + lyr * DMODEL, ln_b + lyr * DMODEL, ws_hnbf, NTOK);

    tdm_warm_kernel<<<1, 32, 0, stream>>>(
        wb_in + (size_t)lyr * 2 * DI * DMODEL, tdm_sink + lyr);

    gemm(ws_hnbf, wb_in + (size_t)lyr * 2 * DI * DMODEL, nullptr,
         ws_xz, NTOK, 2 * DI, DMODEL);

    conv_silu_kernel<<<(NTOK * DI + 255) / 256, 256, 0, stream>>>(
        ws_xz, conv_w + (size_t)lyr * DI * KCONV, conv_b + (size_t)lyr * DI,
        ws_xc, ws_xcbf);

    gemm(ws_xcbf, wb_xp + (size_t)lyr * XPN * DI, nullptr,
         ws_xdbl, NTOK, XPN, DI);

    scan_kernel<<<dim3(3, BATCH), 128, 0, stream>>>(
        ws_xdbl, ws_xc, ws_xz,
        dt_w + (size_t)lyr * DI * DTR, dt_b + (size_t)lyr * DI,
        A_log + (size_t)lyr * DI * NSTATE, D_param + (size_t)lyr * DI, ws_ybf);

    gemm(ws_ybf, wb_out + (size_t)lyr * DMODEL * DI, nullptr,
         ws_hid, NTOK, DMODEL, DI);
  }

  // -------- final LN (last token) + classifier head --------
  final_ln_kernel<<<(BATCH + 7) / 8, 256, 0, stream>>>(
      ws_res, ws_hid, normf_w, normf_b, ws_hd_in);
  gemm(ws_hd_in, wb_head, head_b, out, BATCH, NCLS, DMODEL);
}
